// CGCN_15607911153833
// MI455X (gfx1250) — matrix operand order repaired and checked by reading, not compiled, not run on hardware
//
#include <hip/hip_runtime.h>
#include <hip/hip_bf16.h>

// ---------- problem constants ----------
#define B_  4
#define T_  2048
#define D_  512
#define H_  8
#define DK  64
#define BTD ((size_t)B_ * T_ * D_)
#define DD  ((size_t)D_ * D_)
#define MASK_WITH  (-1.0e10f)
#define LN_EPS     (1e-5f)
#define SCALE      (0.125f)   // 1/sqrt(64)
#define NKT        (T_ / 16)  // 128 key tiles

// gfx1250 async global->LDS path (guarded: falls back to direct loads)
#if defined(__gfx1250__) && \
    __has_builtin(__builtin_amdgcn_global_load_async_to_lds_b128) && \
    __has_builtin(__builtin_amdgcn_s_wait_asynccnt)
#define ASYNC_LDS 1
#else
#define ASYNC_LDS 0
#endif

typedef __bf16 v16bf __attribute__((ext_vector_type(16)));
typedef __bf16 v4bf  __attribute__((ext_vector_type(4)));
typedef float  v8f   __attribute__((ext_vector_type(8)));

#if ASYNC_LDS
// exact pointee type expected by the async builtins: int __vector_size__(16)
typedef int async_v4i __attribute__((vector_size(16)));
typedef __attribute__((address_space(1))) async_v4i* async_gptr;
typedef __attribute__((address_space(3))) async_v4i* async_lptr;
#endif

// D = A(16x32 bf16) * B(32x16 bf16) + C(16x16 f32)
__device__ __forceinline__ v8f wmma_bf16(v16bf a, v16bf b, v8f c) {
    return __builtin_amdgcn_wmma_f32_16x16x32_bf16(false, a, false, b,
                                                   (short)0, c, false, false);
}

// A-fragment: lane = M row; K values kbase+0..7 and kbase+16..23,
// kbase = kk + (lane>>4)*8.
__device__ __forceinline__ v16bf a_frag_bf16(const __bf16* p) {
    v16bf a;
#pragma unroll
    for (int i = 0; i < 8; ++i) a[i]     = p[i];
#pragma unroll
    for (int i = 0; i < 8; ++i) a[8 + i] = p[16 + i];
    return a;
}
// B-fragment: lane = N col; K values kbase+0..7 and kbase+16..23,
// kbase = kk + (lane>>4)*8.
__device__ __forceinline__ v16bf b_frag_bf16(const __bf16* p) {
    v16bf b;
#pragma unroll
    for (int i = 0; i < 8; ++i) b[i]     = p[i];
#pragma unroll
    for (int i = 0; i < 8; ++i) b[8 + i] = p[16 + i];
    return b;
}

// ============================================================================
// Kernel 0: streaming f32 -> bf16 conversion (done ONCE per element, so the
// GEMM kernels load pure bf16 fragments with no cvt dependency chains).
// ============================================================================
__global__ __launch_bounds__(256) void cgcn_cvt(const float* __restrict__ src,
                                                __bf16* __restrict__ dst,
                                                int n4) {
    const int i = blockIdx.x * 256 + threadIdx.x;   // float4 index
    if (i < n4) {
        const float4 v = ((const float4*)src)[i];
        v4bf o;
        o[0] = (__bf16)v.x; o[1] = (__bf16)v.y;
        o[2] = (__bf16)v.z; o[3] = (__bf16)v.w;
        ((v4bf*)dst)[i] = o;
    }
}

// ============================================================================
// Kernel 1: fused projections.  z==0: Qp = q@Wq^T   (layout [b][h][t][dk])
//                               z==1: Kp = k@Wk^T   (layout [b][h][t][dk])
//                               z==2: Gt = k@Wg^T   (layout [b][h][dk][t])
// All operands pre-converted bf16. One wave computes a 16(M) x 64(N) tile;
// per k-step all 4 B-fragments are loaded before the 4 WMMAs for MLP.
// ============================================================================
__global__ __launch_bounds__(32, 1) void cgcn_proj(const __bf16* __restrict__ qbf,
                                                   const __bf16* __restrict__ kbf,
                                                   const __bf16* __restrict__ wqb,
                                                   const __bf16* __restrict__ wkb,
                                                   const __bf16* __restrict__ wgb,
                                                   __bf16* __restrict__ Qp,
                                                   __bf16* __restrict__ Kp,
                                                   __bf16* __restrict__ Gt) {
    const int lane = threadIdx.x, ln = lane & 15, half = lane >> 4;
    const int mt = blockIdx.x, ntg = blockIdx.y, which = blockIdx.z;
    const __bf16* A = (which == 0) ? qbf : kbf;
    const __bf16* W = (which == 0) ? wqb : (which == 1) ? wkb : wgb;

    const size_t arow = (size_t)(mt * 16 + ln) * D_;
    v8f c[4] = {};
#pragma unroll 2
    for (int kk = 0; kk < D_; kk += 32) {
        v16bf af = a_frag_bf16(A + arow + kk + half * 8);
        v16bf bf[4];
#pragma unroll
        for (int j = 0; j < 4; ++j)
            bf[j] = b_frag_bf16(W + (size_t)((ntg * 4 + j) * 16 + ln) * D_ +
                                kk + half * 8);
#pragma unroll
        for (int j = 0; j < 4; ++j)
            c[j] = wmma_bf16(af, bf[j], c[j]);
    }
#pragma unroll
    for (int j = 0; j < 4; ++j) {
        const int n = (ntg * 4 + j) * 16 + ln;   // global output feature
        const int h = n / DK, d = n % DK;
#pragma unroll
        for (int r = 0; r < 8; ++r) {
            const int m = mt * 16 + half * 8 + r;    // global token (b*T + t)
            const int b = m / T_, t = m % T_;
            const __bf16 v = (__bf16)c[j][r];
            if (which == 0)
                Qp[((size_t)(b * H_ + h) * T_ + t) * DK + d] = v;
            else if (which == 1)
                Kp[((size_t)(b * H_ + h) * T_ + t) * DK + d] = v;
            else
                Gt[((size_t)(b * H_ + h) * DK + d) * T_ + t] = v;
        }
    }
}

// ============================================================================
// Kernel 2: masked attention, two-pass softmax, fused P@G, writes adj once.
// One wave handles one (b,h) x 32-row q-tile (2 row-tiles share B-fragments).
// K tiles double-buffered into LDS with async global->LDS copies when present.
// ============================================================================
__global__ __launch_bounds__(32, 1) void cgcn_attn(const __bf16* __restrict__ Qp,
                                                   const __bf16* __restrict__ Kp,
                                                   const __bf16* __restrict__ Gt,
                                                   const int* __restrict__ mask,
                                                   float* __restrict__ adj,
                                                   __bf16* __restrict__ Otmp) {
    __shared__ __bf16 pLds[32 * 32];      // P tile staged C-layout -> A-layout
#if ASYNC_LDS
    __shared__ __bf16 kbuf[2][16 * DK];   // double-buffered K tiles (2KB each)
#endif
    const int lane = threadIdx.x, ln = lane & 15, half = lane >> 4;
    const int qt = blockIdx.x;            // q tile (32 rows) 0..63
    const int bh = blockIdx.y;            // b*H + h
    const int b = bh / H_, h = bh % H_;

    const __bf16* Qb = Qp + (size_t)bh * T_ * DK;
    const __bf16* Kb = Kp + (size_t)bh * T_ * DK;
    const __bf16* Gb = Gt + (size_t)bh * DK * T_;
    const int*    mr = mask + b * T_;

    // Q A-fragments: 2 row-tiles x 2 dk-chunks, held in VGPRs for all passes.
    v16bf aq[2][2];
#pragma unroll
    for (int rt = 0; rt < 2; ++rt)
#pragma unroll
        for (int ch = 0; ch < 2; ++ch)
            aq[rt][ch] = a_frag_bf16(Qb + (size_t)(qt * 32 + rt * 16 + ln) * DK +
                                     ch * 32 + half * 8);

#if ASYNC_LDS
    // Copy one 16x64-bf16 K tile (2048B) into kbuf[kt&1]: 4 wave-wide b128 ops.
    auto stage_k = [&](int kt) {
        const char* g = (const char*)(Kb + (size_t)kt * 16 * DK) + lane * 16;
        async_gptr gp = (async_gptr)(g);
        async_lptr lp = (async_lptr)((char*)(&kbuf[kt & 1][0]) + lane * 16);
        __builtin_amdgcn_global_load_async_to_lds_b128(gp, lp, 0, 0);
        __builtin_amdgcn_global_load_async_to_lds_b128(gp, lp, 512, 0);
        __builtin_amdgcn_global_load_async_to_lds_b128(gp, lp, 1024, 0);
        __builtin_amdgcn_global_load_async_to_lds_b128(gp, lp, 1536, 0);
    };
#endif

    // S(2 x 16q x 16k) += Q K^T for one key tile starting at kp_base.
    auto score_tile = [&](const __bf16* kp_base, v8f (&c)[2]) {
#pragma unroll
        for (int ch = 0; ch < 2; ++ch) {
            v16bf bk = b_frag_bf16(kp_base + (size_t)ln * DK + ch * 32 + half * 8);
            c[0] = wmma_bf16(aq[0][ch], bk, c[0]);
            c[1] = wmma_bf16(aq[1][ch], bk, c[1]);
        }
    };

    // ---- pass 1: online row max / sum(exp) ----
    float mx[2][8], su[2][8];
#pragma unroll
    for (int rt = 0; rt < 2; ++rt)
#pragma unroll
        for (int r = 0; r < 8; ++r) { mx[rt][r] = -INFINITY; su[rt][r] = 0.f; }

#if ASYNC_LDS
    stage_k(0);
#endif
    for (int kt = 0; kt < NKT; ++kt) {
#if ASYNC_LDS
        if (kt + 1 < NKT) { stage_k(kt + 1); __builtin_amdgcn_s_wait_asynccnt(4); }
        else              { __builtin_amdgcn_s_wait_asynccnt(0); }
        __syncthreads();
        const __bf16* kp_base = &kbuf[kt & 1][0];
#else
        __builtin_prefetch(Kb + (size_t)((kt + 1) * 16 + ln) * DK, 0, 0);
        const __bf16* kp_base = Kb + (size_t)kt * 16 * DK;
#endif
        v8f c[2] = {};
        score_tile(kp_base, c);
        const bool on = mr[kt * 16 + ln] != 0;
#pragma unroll
        for (int rt = 0; rt < 2; ++rt)
#pragma unroll
            for (int r = 0; r < 8; ++r) {
                float v  = on ? c[rt][r] * SCALE : MASK_WITH;
                float mn = fmaxf(mx[rt][r], v);
                su[rt][r] = su[rt][r] * __expf(mx[rt][r] - mn) + __expf(v - mn);
                mx[rt][r] = mn;
            }
    }
    // reduce across the 16 lanes of each half-wave (rows live per-half)
#pragma unroll
    for (int off = 1; off < 16; off <<= 1) {
#pragma unroll
        for (int rt = 0; rt < 2; ++rt)
#pragma unroll
            for (int r = 0; r < 8; ++r) {
                float mo = __shfl_xor(mx[rt][r], off, 16);
                float so = __shfl_xor(su[rt][r], off, 16);
                float mn = fmaxf(mx[rt][r], mo);
                su[rt][r] = su[rt][r] * __expf(mx[rt][r] - mn) + so * __expf(mo - mn);
                mx[rt][r] = mn;
            }
    }

    // ---- pass 2: emit P to adj (HBM, written once) + accumulate O = P@G ----
    v8f o[2][4] = {};
#if ASYNC_LDS
    stage_k(0);
#endif
    for (int kp2 = 0; kp2 < NKT / 2; ++kp2) {
#pragma unroll
        for (int sub = 0; sub < 2; ++sub) {
            const int kt = kp2 * 2 + sub;
#if ASYNC_LDS
            if (kt + 1 < NKT) { stage_k(kt + 1); __builtin_amdgcn_s_wait_asynccnt(4); }
            else              { __builtin_amdgcn_s_wait_asynccnt(0); }
            __syncthreads();
            const __bf16* kp_base = &kbuf[kt & 1][0];
#else
            const __bf16* kp_base = Kb + (size_t)kt * 16 * DK;
#endif
            v8f c[2] = {};
            score_tile(kp_base, c);
            const int kcol = kt * 16 + ln;
            const bool on = mr[kcol] != 0;
#pragma unroll
            for (int rt = 0; rt < 2; ++rt)
#pragma unroll
                for (int r = 0; r < 8; ++r) {
                    const int row = rt * 16 + half * 8 + r;    // row within 32-block
                    float v = on ? c[rt][r] * SCALE : MASK_WITH;
                    float p = __expf(v - mx[rt][r]) / su[rt][r];
                    adj[((size_t)bh * T_ + (qt * 32 + row)) * T_ + kcol] = p;
                    pLds[row * 32 + sub * 16 + ln] = (__bf16)p;
                }
        }
        __syncthreads();
        v16bf ap[2];
#pragma unroll
        for (int rt = 0; rt < 2; ++rt)
            ap[rt] = a_frag_bf16(&pLds[(rt * 16 + ln) * 32 + half * 8]);
#pragma unroll
        for (int dt = 0; dt < 4; ++dt) {
            const __bf16* g = Gb + (size_t)(dt * 16 + ln) * T_ + kp2 * 32 + half * 8;
            v16bf bg = b_frag_bf16(g);
            o[0][dt] = wmma_bf16(ap[0], bg, o[0][dt]);
            o[1][dt] = wmma_bf16(ap[1], bg, o[1][dt]);
        }
        __syncthreads();
    }

    // epilogue: relu, store merged-head layout (B,T,D) as bf16 for next GEMM
#pragma unroll
    for (int rt = 0; rt < 2; ++rt)
#pragma unroll
        for (int dt = 0; dt < 4; ++dt)
#pragma unroll
            for (int r = 0; r < 8; ++r) {
                const int q = qt * 32 + rt * 16 + half * 8 + r;
                const int d = dt * 16 + ln;
                Otmp[((size_t)(b * T_ + q)) * D_ + h * DK + d] =
                    (__bf16)fmaxf(o[rt][dt][r], 0.f);
            }
}

// ============================================================================
// Kernel 3: tmp2 = relu(attn_out) @ Wo^T + bo   (relu already applied; Wo bf16)
// One wave computes 16(M) x 64(N); all 4 B-fragments loaded before the WMMAs.
// ============================================================================
__global__ __launch_bounds__(32, 1) void cgcn_ogemm(const __bf16* __restrict__ Otmp,
                                                    const __bf16* __restrict__ wob,
                                                    const float* __restrict__ bo,
                                                    float* __restrict__ tmp2) {
    const int lane = threadIdx.x, ln = lane & 15, half = lane >> 4;
    const int mt = blockIdx.x, ntg = blockIdx.y;
    const size_t arow = (size_t)(mt * 16 + ln) * D_;
    v8f c[4] = {};
#pragma unroll 2
    for (int kk = 0; kk < D_; kk += 32) {
        v16bf af = a_frag_bf16(Otmp + arow + kk + half * 8);
        v16bf bf[4];
#pragma unroll
        for (int j = 0; j < 4; ++j)
            bf[j] = b_frag_bf16(wob + (size_t)((ntg * 4 + j) * 16 + ln) * D_ +
                                kk + half * 8);
#pragma unroll
        for (int j = 0; j < 4; ++j)
            c[j] = wmma_bf16(af, bf[j], c[j]);
    }
#pragma unroll
    for (int j = 0; j < 4; ++j) {
        const int n = (ntg * 4 + j) * 16 + ln;
        const float bias = bo[n];
#pragma unroll
        for (int r = 0; r < 8; ++r) {
            const int m = mt * 16 + half * 8 + r;
            tmp2[(size_t)m * D_ + n] = c[j][r] + bias;
        }
    }
}

// ============================================================================
// Kernel 4: per-token  LN -> relu -> (+query) -> LN
// ============================================================================
__device__ __forceinline__ float block_reduce_sum(float v, float* sh) {
#pragma unroll
    for (int off = 16; off > 0; off >>= 1) v += __shfl_xor(v, off, 32);
    __syncthreads();
    if ((threadIdx.x & 31) == 0) sh[threadIdx.x >> 5] = v;
    __syncthreads();
    float total = 0.f;
#pragma unroll
    for (int i = 0; i < 8; ++i) total += sh[i];
    return total;
}

__global__ __launch_bounds__(256) void cgcn_ln(const float* __restrict__ tmp2,
                                               const float* __restrict__ query,
                                               float* __restrict__ out) {
    __shared__ float sh[8];
    const int m = blockIdx.x, t = threadIdx.x;
    const float* row = tmp2 + (size_t)m * D_;
    float x0 = row[t], x1 = row[t + 256];

    float s  = block_reduce_sum(x0 + x1, sh);
    float sq = block_reduce_sum(x0 * x0 + x1 * x1, sh);
    float mean = s * (1.f / D_);
    float inv  = rsqrtf(sq * (1.f / D_) - mean * mean + LN_EPS);
    float y0 = fmaxf((x0 - mean) * inv, 0.f) + query[(size_t)m * D_ + t];
    float y1 = fmaxf((x1 - mean) * inv, 0.f) + query[(size_t)m * D_ + t + 256];

    s  = block_reduce_sum(y0 + y1, sh);
    sq = block_reduce_sum(y0 * y0 + y1 * y1, sh);
    mean = s * (1.f / D_);
    inv  = rsqrtf(sq * (1.f / D_) - mean * mean + LN_EPS);
    out[(size_t)m * D_ + t]       = (y0 - mean) * inv;
    out[(size_t)m * D_ + t + 256] = (y1 - mean) * inv;
}

// ============================================================================
extern "C" void kernel_launch(void* const* d_in, const int* in_sizes, int n_in,
                              void* d_out, int out_size, void* d_ws, size_t ws_size,
                              hipStream_t stream) {
    (void)in_sizes; (void)n_in; (void)out_size; (void)ws_size;
    const float* query = (const float*)d_in[0];
    const float* key   = (const float*)d_in[1];
    const int*   mask  = (const int*)d_in[2];
    const float* Wq    = (const float*)d_in[3];
    const float* Wk    = (const float*)d_in[4];
    const float* Wg    = (const float*)d_in[5];
    const float* Wo    = (const float*)d_in[6];
    const float* bo    = (const float*)d_in[7];

    float* out = (float*)d_out;                       // (B,T,D) f32
    float* adj = out + BTD;                           // (B,H,T,T) f32

    // workspace layout (66MB total):
    __bf16* qbf  = (__bf16*)d_ws;        // 8MB  bf16 query
    __bf16* kbf  = qbf + BTD;            // 8MB  bf16 key
    __bf16* wqb  = kbf + BTD;            // .5MB bf16 Wq
    __bf16* wkb  = wqb + DD;             // .5MB bf16 Wk
    __bf16* wgb  = wkb + DD;             // .5MB bf16 Wg
    __bf16* wob  = wgb + DD;             // .5MB bf16 Wo
    __bf16* Qp   = wob + DD;             // 8MB  [b][h][t][dk]
    __bf16* Kp   = Qp + BTD;             // 8MB  [b][h][t][dk]
    __bf16* Gt   = Kp + BTD;             // 8MB  [b][h][dk][t]
    __bf16* Otmp = Gt + BTD;             // 8MB  (B,T,D) relu(attn out)
    float*  tmp2 = (float*)(Otmp + BTD); // 16MB (B,T,D) pre-LN

    const int n4_btd = (int)(BTD / 4), n4_dd = (int)(DD / 4);
    cgcn_cvt<<<(n4_btd + 255) / 256, 256, 0, stream>>>(query, qbf, n4_btd);
    cgcn_cvt<<<(n4_btd + 255) / 256, 256, 0, stream>>>(key,   kbf, n4_btd);
    cgcn_cvt<<<(n4_dd  + 255) / 256, 256, 0, stream>>>(Wq,    wqb, n4_dd);
    cgcn_cvt<<<(n4_dd  + 255) / 256, 256, 0, stream>>>(Wk,    wkb, n4_dd);
    cgcn_cvt<<<(n4_dd  + 255) / 256, 256, 0, stream>>>(Wg,    wgb, n4_dd);
    cgcn_cvt<<<(n4_dd  + 255) / 256, 256, 0, stream>>>(Wo,    wob, n4_dd);

    cgcn_proj <<<dim3((B_ * T_) / 16, D_ / 64, 3), 32, 0, stream>>>(
        qbf, kbf, wqb, wkb, wgb, Qp, Kp, Gt);
    cgcn_attn <<<dim3(T_ / 32, B_ * H_), 32, 0, stream>>>(
        Qp, Kp, Gt, mask, adj, Otmp);
    cgcn_ogemm<<<dim3((B_ * T_) / 16, D_ / 64), 32, 0, stream>>>(
        Otmp, wob, bo, tmp2);
    cgcn_ln   <<<B_ * T_, 256, 0, stream>>>(tmp2, query, out);
}
